// GCN_LPA_1168231104589
// MI455X (gfx1250) — matrix operand order repaired and checked by reading, not compiled, not run on hardware
//
#include <hip/hip_runtime.h>

typedef __attribute__((ext_vector_type(16))) _Float16 v16h;
typedef __attribute__((ext_vector_type(8)))  _Float16 v8h;
typedef __attribute__((ext_vector_type(8)))  float    v8f;

static constexpr int N_NODES = 100000;
static constexpr int E_EDGES = 1600000;
static constexpr int IN_F    = 128;
static constexpr int HID_F   = 64;
static constexpr int OUT_F   = 40;
static constexpr int M_TILES = N_NODES / 16;   // 6250, exact

// ---------------- utility fills / converts ----------------
__global__ void fill_val_kernel(float* __restrict__ p, int n, float v) {
    int i = blockIdx.x * blockDim.x + threadIdx.x;
    if (i < n) p[i] = v;
}

__global__ void fill_bias_kernel(float* __restrict__ p, int n, int F,
                                 const float* __restrict__ bias) {
    int i = blockIdx.x * blockDim.x + threadIdx.x;
    if (i < n) p[i] = bias[i % F];
}

__global__ void cvt_f32_to_f16_kernel(const float* __restrict__ in,
                                      _Float16* __restrict__ out, int n) {
    int i = blockIdx.x * blockDim.x + threadIdx.x;
    if (i < n) out[i] = (_Float16)in[i];
}

// ReLU fused with f16 convert: h is only consumed by GEMM2 (f16 A operand).
__global__ void relu_cvt_kernel(const float* __restrict__ in,
                                _Float16* __restrict__ out, int n) {
    int i = blockIdx.x * blockDim.x + threadIdx.x;
    if (i < n) out[i] = (_Float16)fmaxf(in[i], 0.0f);
}

// Pack W[K, NCOL] (fp32, row-major) into WMMA B-fragment layout, f16,
// zero-padded to NT*16 columns:
//   Wp[(((nt*KS + ks)*32 + lane)*16 + i] = W[(ks*32 + 8*(lane>>4) + (i&7) + 16*(i>>3))*NCOL + nt*16 + lane%16]
template <int KS, int NT, int NCOL>
__global__ void pack_w_kernel(const float* __restrict__ W, _Float16* __restrict__ Wp) {
    int idx = blockIdx.x * blockDim.x + threadIdx.x;
    constexpr int TOTAL = NT * KS * 32 * 16;
    if (idx >= TOTAL) return;
    int i    = idx & 15;
    int lane = (idx >> 4) & 31;
    int ks   = (idx >> 9) % KS;
    int nt   = (idx >> 9) / KS;
    int k = ks * 32 + ((lane >> 4) * 8) + (i & 7) + ((i >> 3) * 16);
    int n = nt * 16 + (lane & 15);
    Wp[idx] = (n < NCOL) ? (_Float16)W[k * NCOL + n] : (_Float16)0.0f;
}

// ---------------- degree + edge normalization ----------------
__global__ void seg_deg_kernel(const int* __restrict__ row,
                               const float* __restrict__ attr,
                               float* __restrict__ deg) {
    int e = blockIdx.x * blockDim.x + threadIdx.x;
    if (e < E_EDGES) atomicAdd(&deg[row[e]], attr[e]);
}

__global__ void norm_kernel(const int* __restrict__ row,
                            const float* __restrict__ attr,
                            const float* __restrict__ deg,
                            float* __restrict__ nrm) {
    int e = blockIdx.x * blockDim.x + threadIdx.x;
    if (e < E_EDGES) {
        float d = deg[row[e]];
        nrm[e] = attr[e] / (d == 0.0f ? 1.0f : d);
    }
}

// ---------------- SpMM scatter: one thread per (edge, float4 group) ----------
// Index loads amortized over 4 features; b128 gather; 4 f32 atomics/thread.
template <int F>
__global__ void spmm_scatter4_kernel(const int* __restrict__ row,
                                     const int* __restrict__ col,
                                     const float* __restrict__ nrm,
                                     const float* __restrict__ src,
                                     float* __restrict__ dst) {
    constexpr int G = F / 4;
    unsigned tid = blockIdx.x * blockDim.x + threadIdx.x;
    unsigned e = tid / G;
    unsigned g = tid - e * G;
    if (e < (unsigned)E_EDGES) {
        float w = nrm[e];
        const float4 s = *(const float4*)(src + (size_t)col[e] * F + g * 4u);
        float* d = dst + (size_t)row[e] * F + g * 4u;
        atomicAdd(d + 0, w * s.x);
        atomicAdd(d + 1, w * s.y);
        atomicAdd(d + 2, w * s.z);
        atomicAdd(d + 3, w * s.w);
    }
}

// ---------------- GEMM1: Y1[100000,64] = xh[100000,128] @ W1p ----------------
// One wave per M-tile; the wave holds all 4 N-tile accumulators so the A
// fragment is loaded once per K-step and feeds 4 WMMAs (A reuse x4).
__global__ void __launch_bounds__(32)
gemm1_wmma_kernel(const _Float16* __restrict__ xh,
                  const _Float16* __restrict__ Wp,
                  float* __restrict__ Y) {
    const int lane  = threadIdx.x & 31;
    const int mtile = blockIdx.x;
    const int lh = lane >> 4;
    const int ll = lane & 15;

    v8f acc[4] = {};
#pragma unroll
    for (int ks = 0; ks < 4; ++ks) {
        // A fragment: lane holds row M=ll; K = ks*32 + 8*lh + {0..7, 16..23}
        const _Float16* xr = xh + (mtile * 16 + ll) * IN_F + ks * 32 + lh * 8;
        v8h alo = *(const v8h*)xr;
        v8h ahi = *(const v8h*)(xr + 16);
        v16h a = __builtin_shufflevector(alo, ahi,
                 0, 1, 2, 3, 4, 5, 6, 7, 8, 9, 10, 11, 12, 13, 14, 15);
        // Packed B fragments: contiguous 32B per lane, nt stride = 128 v16h
        const v16h* bp = (const v16h*)Wp + (ks * 32 + lane);
#pragma unroll
        for (int nt = 0; nt < 4; ++nt) {
            v16h b = bp[nt * 128];
            acc[nt] = __builtin_amdgcn_wmma_f32_16x16x32_f16(
                false, a, false, b, (short)0, acc[nt], false, false);
        }
    }
#pragma unroll
    for (int nt = 0; nt < 4; ++nt) {
        float* yp = Y + (mtile * 16 + lh * 8) * HID_F + nt * 16 + ll;
#pragma unroll
        for (int v = 0; v < 8; ++v) yp[v * HID_F] = acc[nt][v];
    }
}

// ---------------- GEMM2: Y2[100000,40] = Hh[100000,64] @ W2p ----------------
// N padded to 48 in the packed B (zeros); EXEC all-1s through the WMMAs,
// masking only at the store.
__global__ void __launch_bounds__(32)
gemm2_wmma_kernel(const _Float16* __restrict__ Hh,
                  const _Float16* __restrict__ Wp,
                  float* __restrict__ Y) {
    const int lane  = threadIdx.x & 31;
    const int mtile = blockIdx.x;
    const int lh = lane >> 4;
    const int ll = lane & 15;

    v8f acc[3] = {};
#pragma unroll
    for (int ks = 0; ks < 2; ++ks) {
        const _Float16* hr = Hh + (mtile * 16 + ll) * HID_F + ks * 32 + lh * 8;
        v8h alo = *(const v8h*)hr;
        v8h ahi = *(const v8h*)(hr + 16);
        v16h a = __builtin_shufflevector(alo, ahi,
                 0, 1, 2, 3, 4, 5, 6, 7, 8, 9, 10, 11, 12, 13, 14, 15);
        const v16h* bp = (const v16h*)Wp + (ks * 32 + lane);
#pragma unroll
        for (int nt = 0; nt < 3; ++nt) {
            v16h b = bp[nt * 64];
            acc[nt] = __builtin_amdgcn_wmma_f32_16x16x32_f16(
                false, a, false, b, (short)0, acc[nt], false, false);
        }
    }
#pragma unroll
    for (int nt = 0; nt < 3; ++nt) {
        int n = nt * 16 + ll;
        if (n < OUT_F) {
            float* yp = Y + (mtile * 16 + lh * 8) * OUT_F + n;
#pragma unroll
            for (int v = 0; v < 8; ++v) yp[v * OUT_F] = acc[nt][v];
        }
    }
}

// ---------------- driver ----------------
extern "C" void kernel_launch(void* const* d_in, const int* in_sizes, int n_in,
                              void* d_out, int out_size, void* d_ws, size_t ws_size,
                              hipStream_t stream) {
    const float* x    = (const float*)d_in[0];   // [N,128]
    const float* soft = (const float*)d_in[1];   // [N,40]
    const int*   eidx = (const int*)  d_in[2];   // [2,E]
    const float* attr = (const float*)d_in[3];   // [E]
    const float* W1   = (const float*)d_in[4];   // [128,64]
    const float* b1   = (const float*)d_in[5];   // [64]
    const float* W2   = (const float*)d_in[6];   // [64,40]
    const float* b2   = (const float*)d_in[7];   // [40]

    const int* row = eidx;
    const int* col = eidx + E_EDGES;

    float* out    = (float*)d_out;                   // [N,40]
    float* labels = out + (size_t)N_NODES * OUT_F;   // [N,40]

    char* w = (char*)d_ws;
    float* deg  = (float*)w;  w += (size_t)N_NODES * 4;            // [N]
    float* nrm  = (float*)w;  w += (size_t)E_EDGES * 4;            // [E]
    float* bufA = (float*)w;  w += (size_t)N_NODES * HID_F * 4;    // Y1 / Y2 / L2
    float* bufB = (float*)w;  w += (size_t)N_NODES * HID_F * 4;    // H  / L1
    _Float16* xh  = (_Float16*)w;  w += (size_t)N_NODES * IN_F  * 2;
    _Float16* Hh  = (_Float16*)w;  w += (size_t)N_NODES * HID_F * 2;
    _Float16* W1p = (_Float16*)w;  w += 4 * 4 * 32 * 16 * 2;       // KS=4, NT=4
    _Float16* W2p = (_Float16*)w;  w += 2 * 3 * 32 * 16 * 2;       // KS=2, NT=3

    const int T = 256;
    auto blocks = [](long long n, int t) { return (unsigned)((n + t - 1) / t); };

    // 1) deg = segment_sum(attr, row); norm = attr / safe_deg[row]
    fill_val_kernel<<<blocks(N_NODES, T), T, 0, stream>>>(deg, N_NODES, 0.0f);
    seg_deg_kernel<<<blocks(E_EDGES, T), T, 0, stream>>>(row, attr, deg);
    norm_kernel<<<blocks(E_EDGES, T), T, 0, stream>>>(row, attr, deg, nrm);

    // 2) one-time precision conversion + weight packing into B-fragment layout
    cvt_f32_to_f16_kernel<<<blocks((long long)N_NODES * IN_F, T), T, 0, stream>>>(
        x, xh, N_NODES * IN_F);
    pack_w_kernel<4, 4, HID_F><<<blocks(4 * 4 * 32 * 16, T), T, 0, stream>>>(W1, W1p);
    pack_w_kernel<2, 3, OUT_F><<<blocks(2 * 3 * 32 * 16, T), T, 0, stream>>>(W2, W2p);

    // 3) Y1 = x @ W1  (WMMA f16 -> f32, 1 wave per M-tile, 4 N-tiles/wave)
    gemm1_wmma_kernel<<<M_TILES, 32, 0, stream>>>(xh, W1p, bufA);

    // 4) H = spmm(Y1) + b1 ; ReLU fused with f16 convert for GEMM2
    fill_bias_kernel<<<blocks((long long)N_NODES * HID_F, T), T, 0, stream>>>(
        bufB, N_NODES * HID_F, HID_F, b1);
    spmm_scatter4_kernel<HID_F>
        <<<blocks((long long)E_EDGES * (HID_F / 4), T), T, 0, stream>>>(
            row, col, nrm, bufA, bufB);
    relu_cvt_kernel<<<blocks((long long)N_NODES * HID_F, T), T, 0, stream>>>(
        bufB, Hh, N_NODES * HID_F);

    // 5) Y2 = h @ W2  (WMMA), reuse bufA as [N,40]
    gemm2_wmma_kernel<<<M_TILES, 32, 0, stream>>>(Hh, W2p, bufA);

    // 6) out = spmm(Y2) + b2
    fill_bias_kernel<<<blocks((long long)N_NODES * OUT_F, T), T, 0, stream>>>(
        out, N_NODES * OUT_F, OUT_F, b2);
    spmm_scatter4_kernel<OUT_F>
        <<<blocks((long long)E_EDGES * (OUT_F / 4), T), T, 0, stream>>>(
            row, col, nrm, bufA, out);

    // 7) labels = spmm^3(soft_labels): soft -> bufB -> bufA -> labels
    fill_val_kernel<<<blocks((long long)N_NODES * OUT_F, T), T, 0, stream>>>(
        bufB, N_NODES * OUT_F, 0.0f);
    spmm_scatter4_kernel<OUT_F>
        <<<blocks((long long)E_EDGES * (OUT_F / 4), T), T, 0, stream>>>(
            row, col, nrm, soft, bufB);

    fill_val_kernel<<<blocks((long long)N_NODES * OUT_F, T), T, 0, stream>>>(
        bufA, N_NODES * OUT_F, 0.0f);
    spmm_scatter4_kernel<OUT_F>
        <<<blocks((long long)E_EDGES * (OUT_F / 4), T), T, 0, stream>>>(
            row, col, nrm, bufB, bufA);

    fill_val_kernel<<<blocks((long long)N_NODES * OUT_F, T), T, 0, stream>>>(
        labels, N_NODES * OUT_F, 0.0f);
    spmm_scatter4_kernel<OUT_F>
        <<<blocks((long long)E_EDGES * (OUT_F / 4), T), T, 0, stream>>>(
            row, col, nrm, bufA, labels);
}
